// SwiGLUMoE_20023137534919
// MI455X (gfx1250) — matrix operand order repaired and checked by reading, not compiled, not run on hardware
//
#include <hip/hip_runtime.h>
#include <math.h>

typedef __fp16   v2fp16 __attribute__((ext_vector_type(2)));
typedef _Float16 v8h  __attribute__((ext_vector_type(8)));
typedef _Float16 v16h __attribute__((ext_vector_type(16)));
typedef float    v8f  __attribute__((ext_vector_type(8)));

#define N_TOK 4096
#define C_DIM 1024
#define E_NUM 8
#define H_DIM 4096
#define M_TILE 64          // tokens per block
#define HC     64          // H chunk per iteration
#define CKB    128         // C slab staged per inner step
#define AUX_COEF 0.01f

#define XSTR  68           // dwords/row: sX,sB1,sB2 (64 data + 4 pad)
#define B3STR 36           // dwords/row: sB3 (32 data + 4 pad)
#define HSTR  36           // dwords/row: sHid (32 data + 4 pad)

// ---------------- workspace layout (bytes) ----------------
// 0        : expert_count   E ints (memset 0 per launch)
// 1024     : expert_list    E*N ints (packed pair = n*2+k)
// 262144   : top_w          N*2 floats
// 524288   : f_partials     512*8 floats
// 540672   : p_partials     512*8 floats
// 1048576  : out_k          N*2*C floats (~33.5 MB)

// single-instruction f32x2 -> packed f16x2 (v_cvt_pk_rtz_f16_f32)
__device__ __forceinline__ unsigned int pkh2(float a, float b) {
    union { v2fp16 h; unsigned int u; } z;
    z.h = __builtin_amdgcn_cvt_pkrtz(a, b);
    return z.u;
}
__device__ __forceinline__ v16h cat8(v8h lo, v8h hi) {
    return __builtin_shufflevector(lo, hi, 0,1,2,3,4,5,6,7,8,9,10,11,12,13,14,15);
}

// =====================================================================
// Kernel 1: router — one wave per token (wave32)
// =====================================================================
__global__ __launch_bounds__(256)
void router_kernel(const float* __restrict__ x,
                   const float* __restrict__ rw,
                   int* __restrict__ cnt,
                   int* __restrict__ list,
                   float* __restrict__ topw,
                   float* __restrict__ fpart,
                   float* __restrict__ ppart)
{
    const int wave = threadIdx.x >> 5;
    const int lane = threadIdx.x & 31;
    const int n = blockIdx.x * 8 + wave;

    const float* xr = x + (size_t)n * C_DIM;
    float acc[E_NUM];
#pragma unroll
    for (int e = 0; e < E_NUM; ++e) acc[e] = 0.0f;

    for (int c = lane; c < C_DIM; c += 32) {
        const float xv = xr[c];
#pragma unroll
        for (int e = 0; e < E_NUM; ++e)
            acc[e] = fmaf(xv, rw[e * C_DIM + c], acc[e]);
    }
#pragma unroll
    for (int e = 0; e < E_NUM; ++e) {
        float v = acc[e];
#pragma unroll
        for (int off = 16; off > 0; off >>= 1) v += __shfl_xor(v, off, 32);
        acc[e] = v;
    }

    __shared__ float s_f[8][E_NUM];
    __shared__ float s_p[8][E_NUM];

    if (lane == 0) {
        float mx = acc[0];
#pragma unroll
        for (int e = 1; e < E_NUM; ++e) mx = fmaxf(mx, acc[e]);
        float p[E_NUM], sum = 0.0f;
#pragma unroll
        for (int e = 0; e < E_NUM; ++e) { p[e] = __expf(acc[e] - mx); sum += p[e]; }
        const float inv = 1.0f / sum;
        int am = 0; float amv = -1.0f;
#pragma unroll
        for (int e = 0; e < E_NUM; ++e) {
            p[e] *= inv;
            s_p[wave][e] = p[e];
            if (p[e] > amv) { amv = p[e]; am = e; }
        }
#pragma unroll
        for (int e = 0; e < E_NUM; ++e) s_f[wave][e] = (e == am) ? 1.0f : 0.0f;

        int i0 = 0; float v0 = p[0];
#pragma unroll
        for (int e = 1; e < E_NUM; ++e) if (p[e] > v0) { v0 = p[e]; i0 = e; }
        int i1 = -1; float v1 = -1.0f;
#pragma unroll
        for (int e = 0; e < E_NUM; ++e) if (e != i0 && p[e] > v1) { v1 = p[e]; i1 = e; }

        const float wsum = v0 + v1;
        topw[n * 2 + 0] = v0 / wsum;
        topw[n * 2 + 1] = v1 / wsum;

        const int pos0 = atomicAdd(&cnt[i0], 1);
        list[i0 * N_TOK + pos0] = n * 2 + 0;
        const int pos1 = atomicAdd(&cnt[i1], 1);
        list[i1 * N_TOK + pos1] = n * 2 + 1;
    }
    __syncthreads();
    if (threadIdx.x < E_NUM) {
        float fs = 0.0f, ps = 0.0f;
#pragma unroll
        for (int w = 0; w < 8; ++w) { fs += s_f[w][threadIdx.x]; ps += s_p[w][threadIdx.x]; }
        fpart[blockIdx.x * E_NUM + threadIdx.x] = fs;
        ppart[blockIdx.x * E_NUM + threadIdx.x] = ps;
    }
}

// =====================================================================
// Kernel 2: aux-loss finalize (fixed-order, deterministic)
// =====================================================================
__global__ void aux_kernel(const float* __restrict__ fpart,
                           const float* __restrict__ ppart,
                           float* __restrict__ out_aux)
{
    __shared__ float sf[E_NUM], sp[E_NUM];
    if (threadIdx.x < E_NUM) {
        float fs = 0.0f, ps = 0.0f;
        for (int b = 0; b < 512; ++b) {
            fs += fpart[b * E_NUM + threadIdx.x];
            ps += ppart[b * E_NUM + threadIdx.x];
        }
        sf[threadIdx.x] = fs / (float)N_TOK;
        sp[threadIdx.x] = ps / (float)N_TOK;
    }
    __syncthreads();
    if (threadIdx.x == 0) {
        float s = 0.0f;
#pragma unroll
        for (int e = 0; e < E_NUM; ++e) s += sf[e] * sp[e];
        out_aux[0] = AUX_COEF * (float)E_NUM * s;
    }
}

// =====================================================================
// Kernel 3: per-expert fused SwiGLU MLP, f16 WMMA 16x16x32
//   grid = (N/64 token-tiles, 8 experts), 512 threads = 16 waves
//   wave (rg, cg): rg = rows rg*16..+16, cg = interleaved 16-col tiles
// =====================================================================
__global__ __launch_bounds__(512, 1)
void moe_gemm_kernel(const float* __restrict__ x,
                     const float* __restrict__ w1,
                     const float* __restrict__ w2,
                     const float* __restrict__ w3,
                     const int* __restrict__ cnt,
                     const int* __restrict__ list,
                     const float* __restrict__ topw,
                     float* __restrict__ outk)
{
    const int e = blockIdx.y;
    const int tile = blockIdx.x;
    const int count = cnt[e];
    if (tile * M_TILE >= count) return;

    // phase-shared LDS pool: [sX | sB1 | sB2] (3*64*68 dw) aliased by sB3 (256*36 dw)
    __shared__ unsigned int pool[3 * 64 * XSTR];    // 52224 B
    __shared__ unsigned int hidp[64 * HSTR];        //  9216 B
    __shared__ int   sPair[M_TILE];
    __shared__ float sW[M_TILE];
    __shared__ int   sTok[M_TILE];

    unsigned int* sX  = pool;
    unsigned int* sB1 = pool + 64 * XSTR;
    unsigned int* sB2 = pool + 2 * 64 * XSTR;
    unsigned int* sB3 = pool;                       // alias (9216 <= 13056 dw)

    const int tid = threadIdx.x;
    if (tid < M_TILE) {
        const int idx = tile * M_TILE + tid;
        if (idx < count) {
            const int pr = list[e * N_TOK + idx];
            sPair[tid] = pr;
            sTok[tid]  = pr >> 1;
            sW[tid]    = topw[pr];
        } else {
            sPair[tid] = -1;
            sTok[tid]  = 0;
            sW[tid]    = 0.0f;
        }
    }
    __syncthreads();

    const float* w1e = w1 + (size_t)e * C_DIM * H_DIM;
    const float* w2e = w2 + (size_t)e * C_DIM * H_DIM;
    const float* w3e = w3 + (size_t)e * H_DIM * C_DIM;

    const int wave = tid >> 5;
    const int lane = tid & 31;
    const int rg   = wave >> 2;     // row group (0..3)
    const int cg   = wave & 3;      // col group (0..3)
    const int lhi  = lane >> 4;     // lane half
    const int ln   = lane & 15;

    const v8f zero = {0.f,0.f,0.f,0.f,0.f,0.f,0.f,0.f};
    v8f acc[16];
#pragma unroll
    for (int t = 0; t < 16; ++t) acc[t] = zero;

    for (int h0 = 0; h0 < H_DIM; h0 += HC) {
        v8f a1 = zero, a2 = zero;   // h1, h2 accum (this wave's 16x16 hidden tile)

        for (int kb = 0; kb < C_DIM; kb += CKB) {
            __syncthreads();
            // ---- stage X[:, kb:kb+128] as f16, natural [m][c] layout ----
            {
                const int m   = tid >> 3;
                const int oct = tid & 7;
                const float* src = x + (size_t)sTok[m] * C_DIM + kb + oct * 16;
                float4 f0 = ((const float4*)src)[0];
                float4 f1 = ((const float4*)src)[1];
                float4 f2 = ((const float4*)src)[2];
                float4 f3 = ((const float4*)src)[3];
                unsigned int* dst = sX + m * XSTR + oct * 8;
                ((uint4*)dst)[0] = make_uint4(pkh2(f0.x,f0.y), pkh2(f0.z,f0.w),
                                              pkh2(f1.x,f1.y), pkh2(f1.z,f1.w));
                ((uint4*)dst)[1] = make_uint4(pkh2(f2.x,f2.y), pkh2(f2.z,f2.w),
                                              pkh2(f3.x,f3.y), pkh2(f3.z,f3.w));
            }
            // ---- stage w1/w2 slab transposed: sB[h][c] (c contiguous as f16) ----
#pragma unroll
            for (int i = 0; i < 2; ++i) {
                const int u = tid + i * 512;
                const int p = u >> 4;       // c pair (0..63)
                const int q = u & 15;       // h quad (0..15)
                const float* r1 = w1e + (size_t)(kb + 2*p) * H_DIM + h0 + 4*q;
                const float* r2 = w2e + (size_t)(kb + 2*p) * H_DIM + h0 + 4*q;
                float4 a = *(const float4*)r1;
                float4 b = *(const float4*)(r1 + H_DIM);
                float4 c = *(const float4*)r2;
                float4 d = *(const float4*)(r2 + H_DIM);
                unsigned int* c1 = sB1 + p;
                unsigned int* c2 = sB2 + p;
                c1[(4*q+0)*XSTR] = pkh2(a.x, b.x);
                c1[(4*q+1)*XSTR] = pkh2(a.y, b.y);
                c1[(4*q+2)*XSTR] = pkh2(a.z, b.z);
                c1[(4*q+3)*XSTR] = pkh2(a.w, b.w);
                c2[(4*q+0)*XSTR] = pkh2(c.x, d.x);
                c2[(4*q+1)*XSTR] = pkh2(c.y, d.y);
                c2[(4*q+2)*XSTR] = pkh2(c.z, d.z);
                c2[(4*q+3)*XSTR] = pkh2(c.w, d.w);
                if (kb + CKB < C_DIM) {
                    __builtin_prefetch(r1 + (size_t)CKB * H_DIM, 0, 1);
                    __builtin_prefetch(r2 + (size_t)CKB * H_DIM, 0, 1);
                }
            }
            __syncthreads();
            // ---- GEMM1: h1/h2 tiles, K over this 128-C slab ----
#pragma unroll
            for (int kk = 0; kk < CKB; kk += 32) {
                const unsigned int* ap = sX + (rg*16 + ln) * XSTR + (kk >> 1) + 8*lhi;
                v16h af = cat8(*(const v8h*)ap, *(const v8h*)(ap + 4));
                const unsigned int* b1p = sB1 + (cg*16 + ln) * XSTR + (kk >> 1) + 8*lhi;
                v16h b1f = cat8(*(const v8h*)b1p, *(const v8h*)(b1p + 4));
                const unsigned int* b2p = sB2 + (cg*16 + ln) * XSTR + (kk >> 1) + 8*lhi;
                v16h b2f = cat8(*(const v8h*)b2p, *(const v8h*)(b2p + 4));
                a1 = __builtin_amdgcn_wmma_f32_16x16x32_f16(false, af, false, b1f,
                                                            (short)0, a1, false, false);
                a2 = __builtin_amdgcn_wmma_f32_16x16x32_f16(false, af, false, b2f,
                                                            (short)0, a2, false, false);
            }
        }

        // ---- SwiGLU, hidden tile -> LDS (f16, natural [m][h]) ----
        __syncthreads();
        {
            _Float16* hh = (_Float16*)hidp;
#pragma unroll
            for (int r = 0; r < 8; ++r) {
                const float v1 = a1[r];
                const float s  = v1 / (1.0f + __expf(-v1));
                const float hv = s * a2[r];
                const int m = rg*16 + r + 8*lhi;
                hh[m * (2*HSTR) + cg*16 + ln] = (_Float16)hv;
            }
        }
        __syncthreads();

        // ---- GEMM2: out += hidden(64x64) @ w3[h0:h0+64, :] ----
#pragma unroll
        for (int ccb = 0; ccb < 4; ++ccb) {
            // stage w3 slab transposed: sB3[c][h] (h contiguous as f16)
#pragma unroll
            for (int i = 0; i < 4; ++i) {
                const int u  = tid + i * 512;
                const int up = u >> 6;      // h pair (0..31)
                const int qc = u & 63;      // c quad (0..63)
                const float* r0 = w3e + (size_t)(h0 + 2*up) * C_DIM + ccb*256 + 4*qc;
                float4 a = *(const float4*)r0;
                float4 b = *(const float4*)(r0 + C_DIM);
                unsigned int* c0 = sB3 + up;
                c0[(4*qc+0)*B3STR] = pkh2(a.x, b.x);
                c0[(4*qc+1)*B3STR] = pkh2(a.y, b.y);
                c0[(4*qc+2)*B3STR] = pkh2(a.z, b.z);
                c0[(4*qc+3)*B3STR] = pkh2(a.w, b.w);
            }
            __syncthreads();
#pragma unroll
            for (int kk = 0; kk < HC; kk += 32) {
                const unsigned int* ap = hidp + (rg*16 + ln) * HSTR + (kk >> 1) + 8*lhi;
                v16h af = cat8(*(const v8h*)ap, *(const v8h*)(ap + 4));
#pragma unroll
                for (int tt = 0; tt < 4; ++tt) {
                    const unsigned int* bp = sB3 + (tt*64 + cg*16 + ln) * B3STR
                                             + (kk >> 1) + 8*lhi;
                    v16h bf = cat8(*(const v8h*)bp, *(const v8h*)(bp + 4));
                    acc[ccb*4 + tt] = __builtin_amdgcn_wmma_f32_16x16x32_f16(
                        false, af, false, bf, (short)0, acc[ccb*4 + tt], false, false);
                }
            }
            __syncthreads();    // before next ccb re-stages sB3
        }
    }

    // ---- writeout: scale by routing weight, scatter to pair slot ----
#pragma unroll
    for (int r = 0; r < 8; ++r) {
        const int m  = rg*16 + r + 8*lhi;
        const int pr = sPair[m];
        if (pr >= 0) {
            const float wv = sW[m];
#pragma unroll
            for (int t = 0; t < 16; ++t)
                outk[(size_t)pr * C_DIM + t*64 + cg*16 + ln] = acc[t][r] * wv;
        }
    }
}

// =====================================================================
// Kernel 4: combine the two weighted expert outputs per token
// =====================================================================
__global__ __launch_bounds__(256)
void combine_kernel(const float* __restrict__ outk, float* __restrict__ out)
{
    const size_t g = (size_t)blockIdx.x * blockDim.x + threadIdx.x; // float4 id
    const size_t n  = g >> 8;
    const size_t cc = g & 255;
    const float4* ok = (const float4*)outk;
    const float4 a = ok[(n * 2 + 0) * 256 + cc];
    const float4 b = ok[(n * 2 + 1) * 256 + cc];
    float4 r;
    r.x = a.x + b.x; r.y = a.y + b.y; r.z = a.z + b.z; r.w = a.w + b.w;
    ((float4*)out)[g] = r;
}

// =====================================================================
extern "C" void kernel_launch(void* const* d_in, const int* in_sizes, int n_in,
                              void* d_out, int out_size, void* d_ws, size_t ws_size,
                              hipStream_t stream)
{
    const float* x  = (const float*)d_in[0];
    const float* rw = (const float*)d_in[1];
    const float* w1 = (const float*)d_in[2];
    const float* w2 = (const float*)d_in[3];
    const float* w3 = (const float*)d_in[4];
    float* out = (float*)d_out;

    char* ws = (char*)d_ws;
    int*   cnt   = (int*)(ws + 0);
    int*   list  = (int*)(ws + 1024);
    float* topw  = (float*)(ws + 262144);
    float* fpart = (float*)(ws + 524288);
    float* ppart = (float*)(ws + 524288 + 16384);
    float* outk  = (float*)(ws + 1048576);

    (void)hipMemsetAsync(cnt, 0, E_NUM * sizeof(int), stream);

    router_kernel<<<N_TOK / 8, 256, 0, stream>>>(x, rw, cnt, list, topw, fpart, ppart);
    aux_kernel<<<1, 256, 0, stream>>>(fpart, ppart, out + (size_t)N_TOK * C_DIM);

    dim3 grid(N_TOK / M_TILE, E_NUM);
    moe_gemm_kernel<<<grid, 512, 0, stream>>>(x, w1, w2, w3, cnt, list, topw, outk);

    combine_kernel<<<(N_TOK * C_DIM / 4) / 256, 256, 0, stream>>>(outk, out);
}
